// RecursiveRefinementModel_326417514656
// MI455X (gfx1250) — matrix operand order repaired and checked by reading, not compiled, not run on hardware
//
#include <hip/hip_runtime.h>
#include <hip/hip_bf16.h>

typedef __attribute__((ext_vector_type(16))) _Float16 v16h;
typedef __attribute__((ext_vector_type(2)))  _Float16 v2h;
typedef __attribute__((ext_vector_type(2)))  __fp16   v2fp16;
typedef __attribute__((ext_vector_type(8)))  float    v8f;

// ---------------------------------------------------------------------------
// CDNA5 WMMA 16x16x32 f16 fragment conventions (wave32):
// A (16x32): lane L holds row m=L&15; elem i -> K=16*(L>=16)+i
// B (32x16): lane L holds col n=L&15; elem i -> K=16*(L>=16)+i
// C/D v8f  : elem i -> row = i + 8*(L>=16), col = L&15
// A fragments live in registers (float4 global loads + v_cvt_pk_rtz_f16_f32).
// B fragments are staged in LDS fragment-linear -> one 32B v16h load each.
// OOB rows/cols are *clamped* (their results are discarded in the epilogue);
// only the K dimension is zero-padded (it feeds valid outputs).
// ---------------------------------------------------------------------------

__device__ __forceinline__ v2h pk2(float x, float y)
{
    v2fp16 r = __builtin_amdgcn_cvt_pkrtz(x, y);
    return __builtin_bit_cast(v2h, r);
}

__device__ __forceinline__ void pack8(v16h& dst, int base, float4 u, float4 v)
{
    v2h h;
    h = pk2(u.x, u.y); dst[base + 0] = h[0]; dst[base + 1] = h[1];
    h = pk2(u.z, u.w); dst[base + 2] = h[0]; dst[base + 3] = h[1];
    h = pk2(v.x, v.y); dst[base + 4] = h[0]; dst[base + 5] = h[1];
    h = pk2(v.z, v.w); dst[base + 6] = h[0]; dst[base + 7] = h[1];
}

__device__ __forceinline__ float dot4(float4 u, const float* k)
{
    return u.x * k[0] + u.y * k[1] + u.z * k[2] + u.w * k[3];
}

// A fragment: K runs [kbase+ah, +8) and [kbase+ah+8, +8) from a clamped row ptr
__device__ __forceinline__ v16h load_a_frag(const float* __restrict__ Ap, int kbase, int ah,
                                            int Kd, bool vec)
{
    v16h a;
    if (kbase + 32 <= Kd) {
        if (vec) {
            const float4* p0 = (const float4*)(Ap + kbase + ah);
            const float4* p1 = (const float4*)(Ap + kbase + ah + 8);
            float4 u0 = p0[0], u1 = p0[1], u2 = p1[0], u3 = p1[1];
            pack8(a, 0, u0, u1);
            pack8(a, 8, u2, u3);
        } else {
#pragma unroll
            for (int i = 0; i < 8; i += 2) {
                v2h l = pk2(Ap[kbase + ah + i], Ap[kbase + ah + i + 1]);
                v2h h = pk2(Ap[kbase + ah + 8 + i], Ap[kbase + ah + 8 + i + 1]);
                a[i] = l[0]; a[i + 1] = l[1];
                a[i + 8] = h[0]; a[i + 9] = h[1];
            }
        }
    } else {
#pragma unroll
        for (int i = 0; i < 8; ++i) {
            int k1 = kbase + ah + i, k2 = kbase + ah + 8 + i;
            float x1 = Ap[k1 < Kd ? k1 : 0]; x1 = (k1 < Kd) ? x1 : 0.f;
            float x2 = Ap[k2 < Kd ? k2 : 0]; x2 = (k2 < Kd) ? x2 : 0.f;
            a[i] = (_Float16)x1; a[i + 8] = (_Float16)x2;
        }
    }
    return a;
}

// B fragment, row-contiguous source (TRANSB): 16 contiguous K values
__device__ __forceinline__ v16h load_b_row(const float* __restrict__ Bp, int kbase, int Kd, bool vec)
{
    v16h b;
    if (kbase + 16 <= Kd) {
        if (vec) {
            const float4* p = (const float4*)(Bp + kbase);
            float4 u0 = p[0], u1 = p[1], u2 = p[2], u3 = p[3];
            pack8(b, 0, u0, u1);
            pack8(b, 8, u2, u3);
        } else {
#pragma unroll
            for (int i = 0; i < 16; i += 2) {
                v2h h = pk2(Bp[kbase + i], Bp[kbase + i + 1]);
                b[i] = h[0]; b[i + 1] = h[1];
            }
        }
    } else {
#pragma unroll
        for (int i = 0; i < 16; ++i) {
            int k = kbase + i;
            float x = Bp[k < Kd ? k : 0]; x = (k < Kd) ? x : 0.f;
            b[i] = (_Float16)x;
        }
    }
    return b;
}

// B fragment, column source (normal): 16 K values strided by ldb at clamped col
__device__ __forceinline__ v16h load_b_col(const float* __restrict__ Bm, int kbase, int bnc,
                                           int ldb, int Kd)
{
    v16h b;
    if (kbase + 16 <= Kd) {
        const float* p = Bm + (long long)kbase * ldb + bnc;
#pragma unroll
        for (int i = 0; i < 16; i += 2) {
            v2h h = pk2(p[(long long)i * ldb], p[(long long)(i + 1) * ldb]);
            b[i] = h[0]; b[i + 1] = h[1];
        }
    } else {
#pragma unroll
        for (int i = 0; i < 16; ++i) {
            int k = kbase + i;
            float x = Bm[(long long)(k < Kd ? k : 0) * ldb + bnc];
            x = (k < Kd) ? x : 0.f;
            b[i] = (_Float16)x;
        }
    }
    return b;
}

// Generic WMMA GEMM: Out = act(alpha * A@B (+bias) (+resid))
// flags: 1=TRANSB, 2=BIAS, 4=RESID, 8=SIGMOID.  Batched via blockIdx.z.
__global__ void gemm_wmma(const float* __restrict__ A, const float* __restrict__ Bm,
                          const float* __restrict__ bias, const float* __restrict__ resid,
                          float* __restrict__ Out,
                          int M, int Nd, int Kd, int lda, int ldb, int ldo,
                          long long Abatch, long long Bbatch, long long Obatch,
                          float alpha, int flags)
{
    __shared__ alignas(32) _Float16 Bfrag[2][2048];   // two 32-K slices, 8KB

    const int z = blockIdx.z;
    A   += (long long)z * Abatch;
    Bm  += (long long)z * Bbatch;
    Out += (long long)z * Obatch;
    if (resid) resid += (long long)z * Obatch;

    const int m0 = blockIdx.x * 64;
    const int n0 = blockIdx.y * 64;
    const int t = threadIdx.x;
    const int lane = t & 31, wave = t >> 5;

    const int am = m0 + wave * 16 + (lane & 15);
    const int amc = (am < M) ? am : (M - 1);               // clamped row
    const int ah = (lane >> 4) * 16;
    const int bn = n0 + wave * 16 + (lane & 15);
    const int bnc = (bn < Nd) ? bn : (Nd - 1);             // clamped col
    const int bk = (lane >> 4) * 16;
    const float* Ap = A + (long long)amc * lda;
    const bool avec = ((lda & 3) == 0);
    const bool bvec = ((ldb & 3) == 0);

    v8f acc[4] = {};
    const int ksteps = (Kd + 63) >> 6;

    for (int kt = 0; kt < ksteps; ++kt) {
        const int k0 = kt << 6;
        v16h a0 = load_a_frag(Ap, k0, ah, Kd, avec);
        v16h a1 = load_a_frag(Ap, k0 + 32, ah, Kd, avec);
        v16h b0, b1;
        if (flags & 1) {
            const float* Bp = Bm + (long long)bnc * ldb;
            b0 = load_b_row(Bp, k0 + bk, Kd, bvec);
            b1 = load_b_row(Bp, k0 + 32 + bk, Kd, bvec);
        } else {
            b0 = load_b_col(Bm, k0 + bk, bnc, ldb, Kd);
            b1 = load_b_col(Bm, k0 + 32 + bk, bnc, ldb, Kd);
        }
        if (k0 + 128 <= Kd) __builtin_prefetch(Ap + k0 + 64, 0, 1);   // global_prefetch_b8

        __syncthreads();
        *(v16h*)&Bfrag[0][(wave * 32 + lane) * 16] = b0;
        *(v16h*)&Bfrag[1][(wave * 32 + lane) * 16] = b1;
        __syncthreads();

#pragma unroll
        for (int nt = 0; nt < 4; ++nt) {
            v16h t0 = *(const v16h*)&Bfrag[0][(nt * 32 + lane) * 16];
            acc[nt] = __builtin_amdgcn_wmma_f32_16x16x32_f16(
                false, a0, false, t0, (short)0, acc[nt], false, false);
            v16h t1 = *(const v16h*)&Bfrag[1][(nt * 32 + lane) * 16];
            acc[nt] = __builtin_amdgcn_wmma_f32_16x16x32_f16(
                false, a1, false, t1, (short)0, acc[nt], false, false);
        }
    }

#pragma unroll
    for (int nt = 0; nt < 4; ++nt)
#pragma unroll
        for (int i = 0; i < 8; ++i) {
            int r = m0 + wave * 16 + (lane >> 4) * 8 + i;
            int c = n0 + nt * 16 + (lane & 15);
            if (r < M && c < Nd) {
                float v = acc[nt][i] * alpha;
                if (flags & 2) v += bias[c];
                if (flags & 4) v += resid[(long long)r * ldo + c];
                if (flags & 8) v = 1.f / (1.f + __expf(-v));
                Out[(long long)r * ldo + c] = v;
            }
        }
}

// ---------------------------------------------------------------------------
// Linear attention phase 1: per (b,head): KV[64x64] = kf^T @ vh over N tokens,
// ksum[64] = sum_n kf (accumulated in f32).  Inputs d-major: fkT/fvT[(bh*64+d)*N+n].
// KV written as an f16 fragment-ordered buffer for apply_attn.
// ---------------------------------------------------------------------------
__global__ void kv_reduce(const float* __restrict__ kfT, const float* __restrict__ fvT,
                          _Float16* __restrict__ KVh, float* __restrict__ ksum, int Ntok)
{
    __shared__ alignas(32) _Float16 Bfrag[2][2048];
    const int bh = blockIdx.x;
    kfT += (long long)bh * 64 * Ntok;
    fvT += (long long)bh * 64 * Ntok;
    const int t = threadIdx.x, lane = t & 31, wave = t >> 5;
    const int ad = wave * 16 + (lane & 15);
    const int ah = (lane >> 4) * 16;
    const int be = wave * 16 + (lane & 15);
    const int bk = (lane >> 4) * 16;
    const float* Ap = kfT + (long long)ad * Ntok;
    const float* Bp = fvT + (long long)be * Ntok;

    v8f acc[4] = {};
    float ks = 0.f;
    const int ksteps = (Ntok + 63) >> 6;

    for (int kt = 0; kt < ksteps; ++kt) {
        const int k0 = kt << 6;
        v16h a0, a1, b0, b1;
        if (k0 + 64 <= Ntok) {      // Ntok % 4 == 0 in this model -> aligned float4
            const float4* pa0 = (const float4*)(Ap + k0 + ah);
            const float4* pa1 = (const float4*)(Ap + k0 + ah + 8);
            const float4* pa2 = (const float4*)(Ap + k0 + 32 + ah);
            const float4* pa3 = (const float4*)(Ap + k0 + 32 + ah + 8);
            float4 u0 = pa0[0], u1 = pa0[1], u2 = pa1[0], u3 = pa1[1];
            float4 u4 = pa2[0], u5 = pa2[1], u6 = pa3[0], u7 = pa3[1];
            ks += (u0.x + u0.y + u0.z + u0.w) + (u1.x + u1.y + u1.z + u1.w)
                + (u2.x + u2.y + u2.z + u2.w) + (u3.x + u3.y + u3.z + u3.w)
                + (u4.x + u4.y + u4.z + u4.w) + (u5.x + u5.y + u5.z + u5.w)
                + (u6.x + u6.y + u6.z + u6.w) + (u7.x + u7.y + u7.z + u7.w);
            pack8(a0, 0, u0, u1); pack8(a0, 8, u2, u3);
            pack8(a1, 0, u4, u5); pack8(a1, 8, u6, u7);
            const float4* pb0 = (const float4*)(Bp + k0 + bk);
            const float4* pb1 = (const float4*)(Bp + k0 + 32 + bk);
            float4 w0 = pb0[0], w1 = pb0[1], w2 = pb0[2], w3 = pb0[3];
            float4 w4 = pb1[0], w5 = pb1[1], w6 = pb1[2], w7 = pb1[3];
            pack8(b0, 0, w0, w1); pack8(b0, 8, w2, w3);
            pack8(b1, 0, w4, w5); pack8(b1, 8, w6, w7);
            if (k0 + 128 <= Ntok) __builtin_prefetch(Ap + k0 + 64, 0, 1);
        } else {
#pragma unroll
            for (int i = 0; i < 8; ++i) {
                int k1 = k0 + ah + i, k2 = k0 + ah + 8 + i;
                int k3 = k0 + 32 + ah + i, k4 = k0 + 32 + ah + 8 + i;
                float x1 = Ap[k1 < Ntok ? k1 : 0]; x1 = (k1 < Ntok) ? x1 : 0.f;
                float x2 = Ap[k2 < Ntok ? k2 : 0]; x2 = (k2 < Ntok) ? x2 : 0.f;
                float x3 = Ap[k3 < Ntok ? k3 : 0]; x3 = (k3 < Ntok) ? x3 : 0.f;
                float x4 = Ap[k4 < Ntok ? k4 : 0]; x4 = (k4 < Ntok) ? x4 : 0.f;
                ks += x1 + x2 + x3 + x4;
                a0[i] = (_Float16)x1; a0[i + 8] = (_Float16)x2;
                a1[i] = (_Float16)x3; a1[i + 8] = (_Float16)x4;
            }
#pragma unroll
            for (int i = 0; i < 16; ++i) {
                int kA = k0 + bk + i, kB = k0 + 32 + bk + i;
                float yA = Bp[kA < Ntok ? kA : 0]; yA = (kA < Ntok) ? yA : 0.f;
                float yB = Bp[kB < Ntok ? kB : 0]; yB = (kB < Ntok) ? yB : 0.f;
                b0[i] = (_Float16)yA; b1[i] = (_Float16)yB;
            }
        }
        __syncthreads();
        *(v16h*)&Bfrag[0][(wave * 32 + lane) * 16] = b0;
        *(v16h*)&Bfrag[1][(wave * 32 + lane) * 16] = b1;
        __syncthreads();
#pragma unroll
        for (int nt = 0; nt < 4; ++nt) {
            v16h t0 = *(const v16h*)&Bfrag[0][(nt * 32 + lane) * 16];
            acc[nt] = __builtin_amdgcn_wmma_f32_16x16x32_f16(
                false, a0, false, t0, (short)0, acc[nt], false, false);
            v16h t1 = *(const v16h*)&Bfrag[1][(nt * 32 + lane) * 16];
            acc[nt] = __builtin_amdgcn_wmma_f32_16x16x32_f16(
                false, a1, false, t1, (short)0, acc[nt], false, false);
        }
    }

    {   // the two half-lanes of each row cover disjoint K runs
        float tot = ks + __shfl_xor(ks, 16);
        if (lane < 16) ksum[bh * 64 + ad] = tot;
    }

    // Store KV as B-fragments for the qf@KV GEMM (K=64 -> 2 halves x 4 col tiles)
#pragma unroll
    for (int nt = 0; nt < 4; ++nt)
#pragma unroll
        for (int i = 0; i < 8; ++i) {
            int r = wave * 16 + (lane >> 4) * 8 + i;    // K index of KV
            int c = nt * 16 + (lane & 15);              // col of KV
            int kt2 = r >> 5;
            int l2 = (c & 15) + 16 * ((r >> 3) & 1);
            int e2 = (r & 7) + 8 * ((r >> 4) & 1);
            KVh[(long long)bh * 4096 + ((kt2 * 4 + nt) * 32 + l2) * 16 + e2] = (_Float16)acc[nt][i];
        }
}

// ---------------------------------------------------------------------------
// Linear attention phase 2: att = (qf @ KV) * z, z = 1/(qf.ksum + 1e-6).
// grid (ceil(N/16), B); block 128 = 4 waves = 4 heads.  KV fragments come
// straight from global as 32B v16h loads. Output head-interleaved (B,N,256).
// ---------------------------------------------------------------------------
__global__ void apply_attn(const float* __restrict__ qf, const _Float16* __restrict__ KVh,
                           const float* __restrict__ ksum, float* __restrict__ attout, int Ntok)
{
    __shared__ float zlds[4][16];
    const int b = blockIdx.y;
    const int n0 = blockIdx.x * 16;
    const int lane = threadIdx.x & 31, head = threadIdx.x >> 5;
    const int bh = b * 4 + head;
    const float* qb = qf + (long long)bh * Ntok * 64;
    const _Float16* kvb = KVh + (long long)bh * 4096;

    const int ar = n0 + (lane & 15);
    const int arc = (ar < Ntok) ? ar : (Ntok - 1);         // clamp; OOB rows discarded
    const int ah = (lane >> 4) * 8;
    const float* Ap = qb + (long long)arc * 64;

    const float4* p0 = (const float4*)(Ap + ah);
    const float4* p1 = (const float4*)(Ap + 16 + ah);
    const float4* p2 = (const float4*)(Ap + 32 + ah);
    const float4* p3 = (const float4*)(Ap + 48 + ah);
    float4 u0 = p0[0], u1 = p0[1], u2 = p1[0], u3 = p1[1];
    float4 u4 = p2[0], u5 = p2[1], u6 = p3[0], u7 = p3[1];
    v16h a0, a1;
    pack8(a0, 0, u0, u1); pack8(a0, 8, u2, u3);
    pack8(a1, 0, u4, u5); pack8(a1, 8, u6, u7);

    const float* ks_ = ksum + bh * 64;
    float zp = dot4(u0, ks_ + ah)      + dot4(u1, ks_ + ah + 4)
             + dot4(u2, ks_ + 16 + ah) + dot4(u3, ks_ + 16 + ah + 4)
             + dot4(u4, ks_ + 32 + ah) + dot4(u5, ks_ + 32 + ah + 4)
             + dot4(u6, ks_ + 48 + ah) + dot4(u7, ks_ + 48 + ah + 4);
    float tot = zp + __shfl_xor(zp, 16);
    if (lane < 16) zlds[head][lane & 15] = 1.f / (tot + 1e-6f);
    __syncthreads();

    v8f acc[4] = {};
#pragma unroll
    for (int nt = 0; nt < 4; ++nt) {
        v16h b0 = *(const v16h*)&kvb[((0 * 4 + nt) * 32 + lane) * 16];
        acc[nt] = __builtin_amdgcn_wmma_f32_16x16x32_f16(
            false, a0, false, b0, (short)0, acc[nt], false, false);
        v16h b1 = *(const v16h*)&kvb[((1 * 4 + nt) * 32 + lane) * 16];
        acc[nt] = __builtin_amdgcn_wmma_f32_16x16x32_f16(
            false, a1, false, b1, (short)0, acc[nt], false, false);
    }

#pragma unroll
    for (int nt = 0; nt < 4; ++nt)
#pragma unroll
        for (int i = 0; i < 8; ++i) {
            int r = (lane >> 4) * 8 + i;
            int n = n0 + r;
            if (n < Ntok) {
                attout[((long long)b * Ntok + n) * 256 + head * 64 + nt * 16 + (lane & 15)] =
                    acc[nt][i] * zlds[head][r];
            }
        }
}

// ---------------------------------------------------------------------------
// Elementwise / small kernels
// ---------------------------------------------------------------------------
__global__ void embed_kernel(const int* __restrict__ di, const int* __restrict__ dou,
                             const int* __restrict__ ti, const float* __restrict__ te,
                             const float* __restrict__ se, float* __restrict__ cur, int Ntok)
{
    int row = blockIdx.x;
    int b = row / Ntok, j = row % Ntok;
    int ts = Ntok - 900;
    int tok;
    if (j < ts) {
        int d = j / 1800, rem = j % 1800;
        int io = rem / 900, s2 = rem % 900;
        tok = io ? dou[(b * 3 + d) * 900 + s2] : di[(b * 3 + d) * 900 + s2];
    } else {
        tok = ti[b * 900 + (j - ts)];
    }
    int c = threadIdx.x;
    cur[(long long)row * 256 + c] = te[tok * 256 + c] + se[c];
}

__global__ void cache_init(const float* __restrict__ slot_emb, const float* __restrict__ lid_emb,
                           float* __restrict__ cache)
{
    int row = blockIdx.x;                 // B*48
    int slot = row % 48, l = slot / 16;
    int c = threadIdx.x;
    if (c >= 145) return;
    float v;
    if (c < 128)       v = slot_emb[slot * 128 + c];
    else if (c == 128) v = 0.f;
    else if (c < 137)  v = lid_emb[l * 8 + (c - 129)];
    else               v = 0.f;
    cache[(long long)row * 145 + c] = v;
}

// qkv -> qf (n-major), fkT/fvT (d-major transposed), with RoPE + elu+1
__global__ void rope_feat(const float* __restrict__ qkv, float* __restrict__ fq,
                          float* __restrict__ fkT, float* __restrict__ fvT, int Ntok, int rows)
{
    long long tid = (long long)blockIdx.x * 256 + threadIdx.x;
    if (tid >= (long long)rows * 128) return;
    int p = (int)(tid & 127);
    long long row = tid >> 7;
    int hh = p >> 5, i = p & 31;
    int n = (int)(row % Ntok), b = (int)(row / Ntok);
    float inv = __powf(10000.f, -(float)i / 32.f);
    float f = (float)n * inv;
    float cs = __cosf(f), sn = __sinf(f);
    const float* base = qkv + row * 768;
    int d0 = hh * 64 + 2 * i;
    float q1 = base[d0], q2 = base[d0 + 1];
    float k1 = base[256 + d0], k2 = base[256 + d0 + 1];
    float v1 = base[512 + d0], v2 = base[512 + d0 + 1];
    float qr1 = q1 * cs - q2 * sn, qr2 = q1 * sn + q2 * cs;
    float kr1 = k1 * cs - k2 * sn, kr2 = k1 * sn + k2 * cs;
    auto elup1 = [](float x) { return x > 0.f ? x + 1.f : __expf(x); };
    long long bh = b * 4 + hh;
    long long oq = (bh * Ntok + n) * 64 + 2 * i;
    fq[oq] = elup1(qr1); fq[oq + 1] = elup1(qr2);
    long long ot = (bh * 64 + 2 * i) * (long long)Ntok + n;
    fkT[ot] = elup1(kr1); fkT[ot + Ntok] = elup1(kr2);
    fvT[ot] = v1;         fvT[ot + Ntok] = v2;
}

__global__ void softmax48(float* __restrict__ s, int rows)
{
    int r = blockIdx.x * 256 + threadIdx.x;
    if (r >= rows) return;
    float* p = s + (long long)r * 48;
    float mx = -1e30f;
#pragma unroll
    for (int i = 0; i < 48; ++i) mx = fmaxf(mx, p[i]);
    float e[48], sum = 0.f;
#pragma unroll
    for (int i = 0; i < 48; ++i) { e[i] = __expf(p[i] - mx); sum += e[i]; }
    float inv = 1.f / sum;
#pragma unroll
    for (int i = 0; i < 48; ++i) p[i] = e[i] * inv;
}

__global__ void rgate_apply(float* __restrict__ h, const float* __restrict__ rd,
                            const float* __restrict__ Wg, const float* __restrict__ bg, int rows)
{
    int wave = threadIdx.x >> 5, lane = threadIdx.x & 31;
    long long row = (long long)blockIdx.x * 8 + wave;
    if (row >= rows) return;
    float* hp = h + row * 256;
    const float* rp = rd + row * 256;
    float acc = 0.f;
#pragma unroll
    for (int j = 0; j < 8; ++j) acc += hp[lane + 32 * j] * Wg[lane + 32 * j];
#pragma unroll
    for (int o = 16; o > 0; o >>= 1) acc += __shfl_xor(acc, o);
    float g = 1.f / (1.f + __expf(-(acc + bg[0])));
#pragma unroll
    for (int j = 0; j < 8; ++j) { int c = lane + 32 * j; hp[c] += g * rp[c]; }
}

// Cache write step: one block per batch, VALU + LDS (f16 wk/wv to fit 64KB)
__global__ void write_step(const float* __restrict__ h, float* __restrict__ cache,
                           const float* __restrict__ Wwk, const float* __restrict__ Wwv,
                           const float* __restrict__ Wwg, const float* __restrict__ bwg,
                           int l, int it_idx, int pas_idx, int Ntok)
{
    __shared__ _Float16 wk[64 * 128];
    __shared__ _Float16 wv[64 * 128];
    __shared__ float sq[16 * 128];
    __shared__ float sc[16 * 64];
    __shared__ float nc[16 * 128];
    __shared__ float wg[16];
    const int b = blockIdx.x, t = threadIdx.x;
    const float* wt = h + ((long long)b * Ntok + (Ntok - 64)) * 256;

    for (int e = t; e < 8192; e += 256) {
        int m = e >> 7, c = e & 127;
        const float* hr = wt + m * 256;
        float a1 = 0.f, a2 = 0.f;
        for (int k = 0; k < 256; ++k) {
            float hv = hr[k];
            a1 += hv * Wwk[k * 128 + c];
            a2 += hv * Wwv[k * 128 + c];
        }
        wk[e] = (_Float16)a1; wv[e] = (_Float16)a2;
    }
    for (int e = t; e < 2048; e += 256) {
        int k = e >> 7, c = e & 127;
        sq[e] = cache[((long long)b * 48 + l * 16 + k) * 145 + c];
    }
    __syncthreads();
    for (int e = t; e < 1024; e += 256) {
        int k = e >> 6, m = e & 63;
        float a = 0.f;
        for (int c = 0; c < 128; ++c) a += sq[k * 128 + c] * (float)wk[m * 128 + c];
        sc[e] = a * 0.08838834764831845f;    // 1/sqrt(128)
    }
    __syncthreads();
    if (t < 16) {
        float mx = -1e30f;
        for (int m = 0; m < 64; ++m) mx = fmaxf(mx, sc[t * 64 + m]);
        float s = 0.f;
        for (int m = 0; m < 64; ++m) { float e2 = __expf(sc[t * 64 + m] - mx); sc[t * 64 + m] = e2; s += e2; }
        float inv = 1.f / s;
        for (int m = 0; m < 64; ++m) sc[t * 64 + m] *= inv;
    }
    __syncthreads();
    for (int e = t; e < 2048; e += 256) {
        int k = e >> 7, c = e & 127;
        float a = 0.f;
        for (int m = 0; m < 64; ++m) a += sc[k * 64 + m] * (float)wv[m * 128 + c];
        nc[e] = a;
    }
    __syncthreads();
    if (t < 16) {
        float a = 0.f;
        for (int c = 0; c < 128; ++c) a += sq[t * 128 + c] * Wwg[c] + nc[t * 128 + c] * Wwg[128 + c];
        wg[t] = 1.f / (1.f + __expf(-(a + bwg[0])));
    }
    __syncthreads();
    for (int e = t; e < 2048; e += 256) {
        int k = e >> 7, c = e & 127;
        cache[((long long)b * 48 + l * 16 + k) * 145 + c] = sq[e] + wg[k] * nc[e];
    }
    if (t < 16) {
        long long base = ((long long)b * 48 + l * 16 + t) * 145;
        cache[base + 128] = wg[t];
        for (int q2 = 0; q2 < 4; ++q2) {
            cache[base + 137 + q2] = (q2 == it_idx) ? 1.f : 0.f;
            cache[base + 141 + q2] = (q2 == pas_idx) ? 1.f : 0.f;
        }
    }
}

__global__ void fb_prepare(const float* __restrict__ logits, const float* __restrict__ cur,
                           const float* __restrict__ pae_tab, float* __restrict__ fbin,
                           float* __restrict__ pbuf, int Ntok)
{
    __shared__ int best;
    int row = blockIdx.x;                 // B*900
    int b = row / 900, s2 = row % 900;
    if (threadIdx.x == 0) {
        const float* lp = logits + (long long)row * 16;
        int bi = 0; float bv = lp[0];
        for (int i = 1; i < 16; ++i) if (lp[i] > bv) { bv = lp[i]; bi = i; }
        best = bi;
    }
    __syncthreads();
    int c = threadIdx.x;
    float tr = cur[((long long)(b * Ntok) + (Ntok - 900) + s2) * 256 + c];
    float pv = pae_tab[best * 256 + c];
    fbin[(long long)row * 512 + c] = tr;
    fbin[(long long)row * 512 + 256 + c] = pv;
    pbuf[(long long)row * 256 + c] = pv;
}

__global__ void fb_apply(float* __restrict__ cur, const float* __restrict__ g,
                         const float* __restrict__ pbuf, int Ntok)
{
    int row = blockIdx.x;
    int b = row / 900, s2 = row % 900;
    int c = threadIdx.x;
    long long ci = ((long long)(b * Ntok) + (Ntok - 900) + s2) * 256 + c;
    cur[ci] += g[(long long)row * 256 + c] * pbuf[(long long)row * 256 + c];
}

// ---------------------------------------------------------------------------
// Host orchestration
// ---------------------------------------------------------------------------
extern "C" void kernel_launch(void* const* d_in, const int* in_sizes, int n_in,
                              void* d_out, int out_size, void* d_ws, size_t ws_size,
                              hipStream_t stream)
{
    (void)in_sizes; (void)n_in; (void)out_size; (void)ws_size;
    const int B = 16, Vv = 16, D = 256, DC = 128, L = 3, H = 4, S = 900;
    const int N = 3 * 2 * S + S;          // 6300
    const int TS = 3 * 2 * S;             // 5400
    const int DSLOT = 145;

    const int* demo_in  = (const int*)d_in[0];
    const int* demo_out = (const int*)d_in[1];
    const int* test_in  = (const int*)d_in[2];
    const float* token_embed = (const float*)d_in[3];
    const float* seg_embed   = (const float*)d_in[4];
    const float* slot_emb    = (const float*)d_in[5];
    const float* lid_emb     = (const float*)d_in[6];
    const float* Wq   = (const float*)d_in[7];
    const float* Wk   = (const float*)d_in[8];
    const float* Wv   = (const float*)d_in[9];
    const float* Wg   = (const float*)d_in[10];
    const float* bg   = (const float*)d_in[11];
    const float* Wqkv = (const float*)d_in[12];
    const float* Wo   = (const float*)d_in[13];
    const float* Wwk  = (const float*)d_in[14];
    const float* Wwv  = (const float*)d_in[15];
    const float* Wwg  = (const float*)d_in[16];
    const float* bwg  = (const float*)d_in[17];
    const float* Wout = (const float*)d_in[18];
    const float* bout = (const float*)d_in[19];
    const float* pae_tab = (const float*)d_in[20];
    const float* Wfb  = (const float*)d_in[21];
    const float* bfb  = (const float*)d_in[22];
    float* out = (float*)d_out;

    float* ws = (float*)d_ws;
    size_t off = 0;
    auto alloc = [&](size_t n) { float* p = ws + off; off += n; return p; };
    const size_t BN = (size_t)B * N;
    float* cur   = alloc(BN * D);
    float* hb    = alloc(BN * D);
    float* qb    = alloc(BN * DC);
    float* scb   = alloc(BN * 48);
    float* rd    = alloc(BN * D);         // read buffer / attout / fb gate
    float* qkvb  = alloc(BN * 3 * D);
    float* fq    = alloc(BN * D);         // (B*H, N, 64) n-major
    float* fkT   = alloc(BN * D);         // (B*H, 64, N) d-major
    float* fvT   = alloc(BN * D);         // (B*H, 64, N) d-major
    float* kc    = alloc((size_t)B * 48 * DC);
    float* vc    = alloc((size_t)B * 48 * D);
    _Float16* KVh = (_Float16*)alloc((size_t)B * H * 64 * 64 / 2);  // f16 fragment-ordered
    float* ksb   = alloc((size_t)B * H * 64);
    float* cacheb= alloc((size_t)B * 48 * DSLOT);
    float* fbin  = qb;                    // alias (B*900*512 <= BN*DC)
    float* pbuf  = scb;                   // alias (B*900*256 <= BN*48)
    float* gfb   = rd;                    // alias

    auto gemm = [&](const float* A, const float* Bm, const float* bias, const float* resid,
                    float* O, int M, int Nd, int Kd, int lda, int ldb, int ldo,
                    long long Ab, long long Bb, long long Ob, int NB,
                    float alpha, int flags) {
        dim3 g((M + 63) / 64, (Nd + 63) / 64, NB);
        gemm_wmma<<<g, 128, 0, stream>>>(A, Bm, bias, resid, O, M, Nd, Kd, lda, ldb, ldo,
                                         Ab, Bb, Ob, alpha, flags);
    };

    embed_kernel<<<B * N, 256, 0, stream>>>(demo_in, demo_out, test_in, token_embed, seg_embed, cur, N);
    cache_init<<<B * 48, 256, 0, stream>>>(slot_emb, lid_emb, cacheb);

    const float rsDC = 0.08838834764831845f;   // 1/sqrt(128)

    for (int pas = 0; pas < 2; ++pas) {
        if (pas) {
            fb_prepare<<<B * S, 256, 0, stream>>>(out, cur, pae_tab, fbin, pbuf, N);
            gemm(fbin, Wfb, bfb, nullptr, gfb, B * S, D, 2 * D, 2 * D, D, D,
                 0, 0, 0, 1, 1.f, 2 | 8);
            fb_apply<<<B * S, 256, 0, stream>>>(cur, gfb, pbuf, N);
        }
        (void)hipMemcpyAsync(hb, cur, BN * D * sizeof(float), hipMemcpyDeviceToDevice, stream);

        for (int l = 0; l < L; ++l)
            for (int it = 0; it < 2; ++it) {
                // ---- slot read attention ----
                gemm(hb, Wq + (size_t)l * D * DC, nullptr, nullptr, qb,
                     B * N, DC, D, D, DC, DC, 0, 0, 0, 1, 1.f, 0);
                gemm(cacheb, Wk + (size_t)l * DSLOT * DC, nullptr, nullptr, kc,
                     B * 48, DC, DSLOT, DSLOT, DC, DC, 0, 0, 0, 1, 1.f, 0);
                gemm(cacheb, Wv + (size_t)l * DSLOT * D, nullptr, nullptr, vc,
                     B * 48, D, DSLOT, DSLOT, D, D, 0, 0, 0, 1, 1.f, 0);
                gemm(qb, kc, nullptr, nullptr, scb, N, 48, DC, DC, DC, 48,
                     (long long)N * DC, 48LL * DC, (long long)N * 48, B, rsDC, 1 /*TRANSB*/);
                softmax48<<<(B * N + 255) / 256, 256, 0, stream>>>(scb, B * N);
                gemm(scb, vc, nullptr, nullptr, rd, N, D, 48, 48, D, D,
                     (long long)N * 48, 48LL * D, (long long)N * D, B, 1.f, 0);
                rgate_apply<<<(B * N + 7) / 8, 256, 0, stream>>>(hb, rd, Wg + (size_t)l * D, bg + l, B * N);

                // ---- linear attention ----
                gemm(hb, Wqkv + (size_t)l * D * 3 * D, nullptr, nullptr, qkvb,
                     B * N, 3 * D, D, D, 3 * D, 3 * D, 0, 0, 0, 1, 1.f, 0);
                rope_feat<<<(B * N * 128 + 255) / 256, 256, 0, stream>>>(qkvb, fq, fkT, fvT, N, B * N);
                kv_reduce<<<B * H, 128, 0, stream>>>(fkT, fvT, KVh, ksb, N);
                apply_attn<<<dim3((N + 15) / 16, B), 128, 0, stream>>>(fq, KVh, ksb, rd, N);
                gemm(rd, Wo + (size_t)l * D * D, nullptr, hb, hb,
                     B * N, D, D, D, D, D, 0, 0, 0, 1, 1.f, 4 /*RESID*/);

                // ---- cache write step ----
                write_step<<<B, 256, 0, stream>>>(hb, cacheb,
                                                  Wwk + (size_t)l * D * DC, Wwv + (size_t)l * D * DC,
                                                  Wwg + (size_t)l * 2 * DC, bwg + l, l, it, pas, N);
            }

        // ---- logits ----
        gemm(hb + (size_t)TS * D, Wout, bout, nullptr, out, S, Vv, D, D, Vv, Vv,
             (long long)N * D, 0, (long long)S * Vv, B, 1.f, 2 /*BIAS*/);
    }
}
